// WindowAttentionGlobal_5076651344140
// MI455X (gfx1250) — matrix operand order repaired and checked by reading, not compiled, not run on hardware
//
#include <hip/hip_runtime.h>
#include <hip/hip_bf16.h>

typedef __attribute__((ext_vector_type(16))) __bf16 v16bf;
typedef __attribute__((ext_vector_type(8)))  __bf16 v8bf;
typedef __attribute__((ext_vector_type(8)))  float  v8f;
typedef __attribute__((ext_vector_type(4)))  float  v4f;

#define NTOK     49
#define NPAD     64
#define CDIM     512
#define NHEAD    16
#define HD       32
#define NWIN     1024
#define VT_PITCH 72      // padded token pitch for V^T (bank-conflict mitigation)

// LDS layout (byte offsets)
#define LDS_X    0                        // X then Xo: 64x512 bf16 = 65536
#define LDS_K    65536                    // K: 64x512 bf16        = 65536
#define LDS_VT   131072                   // V^T: 512x72 bf16      = 73728
#define LDS_P    204800                   // P panels: 8x(64x64)   = 65536
#define LDS_BIAS 270336                   // bias: 169*16 f32      = 10816
#define LDS_TOTAL 281152

// ---- fragment helpers (layouts per CDNA5 ISA §7.12.2, wave32) ----
// A (16-bit, 16x32): lane L -> M=L&15; elems e -> K = kOff+e, kOff = (L<16?0:16) (k order used here, not AMD's table order)
// B (16-bit, 32x16): lane L -> N=L&15; elems e -> K = kOff+e, kOff = (L<16?0:16) (k order used here, same as A)
// C/D (f32, 16x16):  vgpr r -> M = r + (L<16?0:8); N = L&15
__device__ __forceinline__ v16bf frag2(const __bf16* p0, const __bf16* p1) {
  v8bf lo = *(const v8bf*)p0;
  v8bf hi = *(const v8bf*)p1;
  return __builtin_shufflevector(lo, hi, 0,1,2,3,4,5,6,7,8,9,10,11,12,13,14,15);
}
__device__ __forceinline__ v16bf frag16(const __bf16* p) { return frag2(p, p + 8); }
__device__ __forceinline__ v8f wmma_bf16(v16bf a, v16bf b, v8f c) {
  return __builtin_amdgcn_wmma_f32_16x16x32_bf16(false, a, false, b, (short)0, c, false, false);
}

// ---- pre-pass: f32 weights -> bf16, pre-swizzled to B-fragment order -------
// layout: [nTile][kChunk(16)][lane(32)][e(16)], value = W[kc*32 + (L<16?0:16)+e][nt*16 + (L&15)]
__global__ void swizzle_weights(const float* __restrict__ qkv_w,
                                const float* __restrict__ proj_w,
                                __bf16* __restrict__ wKV,
                                __bf16* __restrict__ wP) {
  int idx = blockIdx.x * blockDim.x + threadIdx.x;
  if (idx < 512 * 1024) {
    int e  = idx & 15;
    int L  = (idx >> 4) & 31;
    int kc = (idx >> 9) & 15;
    int nt = idx >> 13;                       // 0..63
    int k  = (kc << 5) + ((L & 16) ? 16 : 0) + e;
    int n  = (nt << 4) + (L & 15);
    wKV[idx] = (__bf16)qkv_w[k * 1024 + n];
  } else {
    int j  = idx - 512 * 1024;                // 0..262143
    int e  = j & 15;
    int L  = (j >> 4) & 31;
    int kc = (j >> 9) & 15;
    int nt = j >> 13;                         // 0..31
    int k  = (kc << 5) + ((L & 16) ? 16 : 0) + e;
    int n  = (nt << 4) + (L & 15);
    wP[j] = (__bf16)proj_w[k * 512 + n];
  }
}

// ---- fused window-attention kernel: 1 block = 1 window --------------------
__global__ void __launch_bounds__(256, 1)
win_attn_kernel(const float* __restrict__ inputs,
                const float* __restrict__ q_global,
                const float* __restrict__ qkv_b,
                const float* __restrict__ bias_table,
                const float* __restrict__ proj_b,
                const __bf16* __restrict__ wKV,
                const __bf16* __restrict__ wP,
                float* __restrict__ out) {
  extern __shared__ char smem[];
  __bf16* Xs  = (__bf16*)(smem + LDS_X);      // X, later reused as Xo
  __bf16* Ks  = (__bf16*)(smem + LDS_K);
  __bf16* Vt  = (__bf16*)(smem + LDS_VT);
  __bf16* Pb  = (__bf16*)(smem + LDS_P);
  float*  Bls = (float*)(smem + LDS_BIAS);

  const int b    = blockIdx.x;
  const int tid  = threadIdx.x;
  const int wave = tid >> 5;
  const int lane = tid & 31;
  const int l15  = lane & 15;
  const int hlf  = lane >> 4;
  const int kA   = hlf * 8;                   // C/D row sub-base
  const int kOff = hlf * 16;                  // B-frag K sub-base

  // ======================== Phase 1: stage X (bf16) + bias table ==========
  for (int i = tid; i < 169 * NHEAD; i += 256) Bls[i] = bias_table[i];
  const float* Xg = inputs + (size_t)b * (NTOK * CDIM);
  for (int i = tid; i < NTOK * CDIM / 4; i += 256) {
    v4f x = *(const v4f*)(Xg + i * 4);
    int row = i >> 7, c = (i & 127) << 2;
    __bf16* d = Xs + row * CDIM + c;
    d[0] = (__bf16)x[0]; d[1] = (__bf16)x[1]; d[2] = (__bf16)x[2]; d[3] = (__bf16)x[3];
  }
  for (int i = tid; i < (NPAD - NTOK) * CDIM / 4; i += 256) {
    int row = NTOK + (i >> 7), c = (i & 127) << 2;
    *(uint2*)(Xs + row * CDIM + c) = make_uint2(0u, 0u);
  }
  __syncthreads();

  // ======================== Phase 2: KV = X @ Wkv (+qkv_b) ================
  // M = 64 tokens (4 tiles), N = 1024 (64 tiles, 8 per wave), K = 512 (16 chunks)
  {
    v8f zero = {};
    for (int mt = 0; mt < 4; ++mt) {
      v8f acc[8];
      #pragma unroll
      for (int q = 0; q < 8; ++q) acc[q] = zero;
      const __bf16* xrow = Xs + (mt * 16 + l15) * CDIM;
      for (int kc = 0; kc < 16; ++kc) {
        v16bf a = frag16(xrow + kc * 32 + kOff);
        #pragma unroll
        for (int q = 0; q < 8; ++q) {
          int nt = wave + (q << 3);
          v16bf bm = frag16(wKV + (size_t)(((nt << 4) + kc) * 32 + lane) * 16);
          acc[q] = wmma_bf16(a, bm, acc[q]);
        }
      }
      #pragma unroll
      for (int q = 0; q < 8; ++q) {
        int nt = wave + (q << 3);
        int n  = (nt << 4) + l15;
        float bias = qkv_b[n];
        #pragma unroll
        for (int r = 0; r < 8; ++r) {
          int m = mt * 16 + r + kA;
          __bf16 hv = (__bf16)(acc[q][r] + bias);
          if (nt < 32) Ks[m * CDIM + n] = hv;                 // K row-major [t][c]
          else         Vt[(n - CDIM) * VT_PITCH + m] = hv;    // V transposed [c][t]
        }
      }
    }
  }
  __syncthreads();

  // ======================== Phase 3: attention (2 heads per wave) =========
  const float* qg = q_global + (size_t)(b >> 4) * (NTOK * CDIM);
  const float scale = 0.17677669529663687f;   // 1/sqrt(32)
  for (int rep = 0; rep < 2; ++rep) {
    const int h = wave * 2 + rep;
    // Q fragments straight from global (L2-resident, reused by 16 windows)
    v16bf qf[4];
    #pragma unroll
    for (int i = 0; i < 4; ++i) {
      int t = i * 16 + l15;
      v16bf f;
      if (t < NTOK) {
        const float* qp = qg + t * CDIM + h * HD;
        v4f a0 = *(const v4f*)(qp + kOff);
        v4f a1 = *(const v4f*)(qp + kOff + 4);
        v4f a2 = *(const v4f*)(qp + kOff + 8);
        v4f a3 = *(const v4f*)(qp + kOff + 12);
        #pragma unroll
        for (int e = 0; e < 4; ++e) {
          f[e]      = (__bf16)(a0[e] * scale);
          f[e + 4]  = (__bf16)(a1[e] * scale);
          f[e + 8]  = (__bf16)(a2[e] * scale);
          f[e + 12] = (__bf16)(a3[e] * scale);
        }
      } else {
        #pragma unroll
        for (int e = 0; e < 16; ++e) f[e] = (__bf16)0.0f;
      }
      qf[i] = f;
    }
    // S = Q K^T   (K dim = hd = 32 -> exactly one WMMA per tile pair)
    v8f S[4][4];
    v8f zero = {};
    #pragma unroll
    for (int j = 0; j < 4; ++j) {
      int t = j * 16 + l15;
      v16bf kf = frag16(Ks + t * CDIM + h * HD + kOff);
      #pragma unroll
      for (int i = 0; i < 4; ++i) S[i][j] = wmma_bf16(qf[i], kf, zero);
    }
    // bias + pad mask + register softmax (rows live in one 16-lane half)
    #pragma unroll
    for (int i = 0; i < 4; ++i) {
      #pragma unroll
      for (int j = 0; j < 4; ++j) {
        int n = j * 16 + l15;
        #pragma unroll
        for (int r = 0; r < 8; ++r) {
          int m = i * 16 + r + kA;
          if (n >= NTOK) S[i][j][r] = -1e30f;
          else {
            int mm = (m < NTOK) ? m : (NTOK - 1);
            int ih = mm / 7, iw = mm - ih * 7;
            int jh = n / 7,  jw = n - jh * 7;
            int idx = (ih - jh + 6) * 13 + (iw - jw + 6);
            S[i][j][r] += Bls[idx * NHEAD + h];
          }
        }
      }
      #pragma unroll
      for (int r = 0; r < 8; ++r) {
        float rmax = fmaxf(fmaxf(S[i][0][r], S[i][1][r]), fmaxf(S[i][2][r], S[i][3][r]));
        rmax = fmaxf(rmax, __shfl_xor(rmax, 1, 32));
        rmax = fmaxf(rmax, __shfl_xor(rmax, 2, 32));
        rmax = fmaxf(rmax, __shfl_xor(rmax, 4, 32));
        rmax = fmaxf(rmax, __shfl_xor(rmax, 8, 32));
        float e0 = __expf(S[i][0][r] - rmax);
        float e1 = __expf(S[i][1][r] - rmax);
        float e2 = __expf(S[i][2][r] - rmax);
        float e3 = __expf(S[i][3][r] - rmax);
        float rs = e0 + e1 + e2 + e3;
        rs += __shfl_xor(rs, 1, 32);
        rs += __shfl_xor(rs, 2, 32);
        rs += __shfl_xor(rs, 4, 32);
        rs += __shfl_xor(rs, 8, 32);
        float inv = __builtin_amdgcn_rcpf(rs);
        S[i][0][r] = e0 * inv; S[i][1][r] = e1 * inv;
        S[i][2][r] = e2 * inv; S[i][3][r] = e3 * inv;
      }
    }
    // spill P to this wave's private LDS panel, re-read as A fragments
    __bf16* Pw = Pb + wave * (64 * 64);
    #pragma unroll
    for (int i = 0; i < 4; ++i)
      #pragma unroll
      for (int j = 0; j < 4; ++j)
        #pragma unroll
        for (int r = 0; r < 8; ++r)
          Pw[(i * 16 + r + kA) * 64 + j * 16 + l15] = (__bf16)S[i][j][r];
    // O = P @ V   (K = 64 keys -> 2 chunks; V^T gives contiguous B frags)
    v8f O[4][2];
    #pragma unroll
    for (int i = 0; i < 4; ++i) { O[i][0] = zero; O[i][1] = zero; }
    #pragma unroll
    for (int kc = 0; kc < 2; ++kc) {
      v16bf vb[2];
      #pragma unroll
      for (int dt = 0; dt < 2; ++dt)
        vb[dt] = frag16(Vt + (h * HD + dt * 16 + l15) * VT_PITCH + kc * 32 + kOff);
      #pragma unroll
      for (int i = 0; i < 4; ++i) {
        const __bf16* prow = Pw + (i * 16 + l15) * 64 + kc * 32;
        v16bf pa = frag16(prow + kOff);
        #pragma unroll
        for (int dt = 0; dt < 2; ++dt) O[i][dt] = wmma_bf16(pa, vb[dt], O[i][dt]);
      }
    }
    // write head output into Xo (reuses X region; disjoint columns per head)
    #pragma unroll
    for (int i = 0; i < 4; ++i)
      #pragma unroll
      for (int dt = 0; dt < 2; ++dt)
        #pragma unroll
        for (int r = 0; r < 8; ++r)
          Xs[(i * 16 + r + kA) * CDIM + h * HD + dt * 16 + l15] = (__bf16)O[i][dt][r];
  }
  __syncthreads();

  // ======================== Phase 4: out = Xo @ Wproj + proj_b ============
  {
    v8f zero = {};
    float* orow = out + (size_t)b * (NTOK * CDIM);
    for (int mt = 0; mt < 4; ++mt) {
      v8f acc[4];
      #pragma unroll
      for (int q = 0; q < 4; ++q) acc[q] = zero;
      const __bf16* xrow = Xs + (mt * 16 + l15) * CDIM;
      for (int kc = 0; kc < 16; ++kc) {
        v16bf a = frag16(xrow + kc * 32 + kOff);
        #pragma unroll
        for (int q = 0; q < 4; ++q) {
          int nt = wave + (q << 3);
          v16bf bm = frag16(wP + (size_t)(((nt << 4) + kc) * 32 + lane) * 16);
          acc[q] = wmma_bf16(a, bm, acc[q]);
        }
      }
      #pragma unroll
      for (int q = 0; q < 4; ++q) {
        int n = ((wave + (q << 3)) << 4) + l15;
        float pb = proj_b[n];
        #pragma unroll
        for (int r = 0; r < 8; ++r) {
          int m = mt * 16 + r + kA;
          if (m < NTOK) orow[m * CDIM + n] = acc[q][r] + pb;
        }
      }
    }
  }
}

extern "C" void kernel_launch(void* const* d_in, const int* in_sizes, int n_in,
                              void* d_out, int out_size, void* d_ws, size_t ws_size,
                              hipStream_t stream) {
  const float* inputs     = (const float*)d_in[0];
  const float* q_global   = (const float*)d_in[1];
  const float* qkv_w      = (const float*)d_in[2];
  const float* qkv_b      = (const float*)d_in[3];
  const float* bias_table = (const float*)d_in[4];
  const float* proj_w     = (const float*)d_in[5];
  const float* proj_b     = (const float*)d_in[6];
  float* out = (float*)d_out;

  __bf16* wKV = (__bf16*)d_ws;              // 512*1024 bf16 = 1 MB
  __bf16* wP  = wKV + 512 * 1024;           // 512*512  bf16 = 0.5 MB

  // allow >64KB dynamic LDS (WGP has 320KB); deterministic, idempotent
  hipFuncSetAttribute((const void*)win_attn_kernel,
                      hipFuncAttributeMaxDynamicSharedMemorySize, LDS_TOTAL);

  // one-time-per-call weight swizzle: (512*1024 + 512*512) elems / 256 thr
  swizzle_weights<<<dim3((512 * 1024 + 512 * 512) / 256), 256, 0, stream>>>(
      qkv_w, proj_w, wKV, wP);

  win_attn_kernel<<<dim3(NWIN), 256, LDS_TOTAL, stream>>>(
      inputs, q_global, qkv_b, bias_table, proj_b, wKV, wP, out);
}
